// TransformerNetwork_52158082843460
// MI455X (gfx1250) — matrix operand order repaired and checked by reading, not compiled, not run on hardware
//
#include <hip/hip_runtime.h>
#include <hip/hip_bf16.h>

// Problem constants (from reference): L=6 blocks, H=8 heads, D=512, N=2048.
#define LNUM 6
#define HNUM 8
#define DDIM 512
#define NEV  2048
#define HD   (HNUM * DDIM)

typedef __bf16 bf16_t;
typedef __bf16 v16bf __attribute__((ext_vector_type(16)));
typedef float  v8f   __attribute__((ext_vector_type(8)));
typedef unsigned int v4u __attribute__((ext_vector_type(4)));
typedef int v4i __attribute__((ext_vector_type(4)));
typedef int v8i __attribute__((ext_vector_type(8)));

union AFrag { v16bf v; uint4 q[2]; };
union CFrag { v8f   v; float f[8]; };

// B-style fragment: 16 consecutive bf16 (one column of B over 16 K values).
static __device__ __forceinline__ v16bf frag_contig(const bf16_t* p) {
  AFrag a;
  a.q[0] = *(const uint4*)(p);
  a.q[1] = *(const uint4*)(p + 8);
  return a.v;
}
// A-style fragment: 16 consecutive K values, K=base..base+15
// (base = 0 for lanes 0-15, 16 for 16-31), matching the B fragments.
static __device__ __forceinline__ v16bf frag_split(const bf16_t* p) {
  AFrag a;
  a.q[0] = *(const uint4*)(p);
  a.q[1] = *(const uint4*)(p + 8);
  return a.v;
}

static __device__ __forceinline__ v8f wmma_bf16(v16bf a, v16bf b, v8f c) {
  return __builtin_amdgcn_wmma_f32_16x16x32_bf16(
      /*neg_a=*/false, a, /*neg_b=*/false, b,
      /*c_mod=*/(short)0, c, /*reuse_a=*/false, /*reuse_b=*/false);
}

// ---------------------------------------------------------------------------
// Staging-path selection: TDM (tensor_load_to_lds) > async global->LDS > plain.
// ---------------------------------------------------------------------------
#if __has_builtin(__builtin_amdgcn_tensor_load_to_lds)
#define STAGE_TDM 1
#elif __has_builtin(__builtin_amdgcn_global_load_async_to_lds_b128)
#define STAGE_ASYNC 1
#endif

// LDS byte offset of a generic pointer into workgroup LDS.
static __device__ __forceinline__ unsigned lds_off(const void* p) {
#if defined(__HIP_DEVICE_COMPILE__)
  return (unsigned)(size_t)(__attribute__((address_space(3))) const void*)p;
#else
  (void)p;
  return 0u;
#endif
}

#if defined(STAGE_TDM)
// Issue a 2-D TDM load: tile (tile0 x tile1) of 2-byte elements from a row-major
// tensor (row pitch stride0 elements) into LDS at lds_addr. D# per ISA §8.3-8.4.
static __device__ __forceinline__ void tdm_load_2d(unsigned lds_addr, const void* gaddr,
                                                   unsigned tile0, unsigned tile1,
                                                   unsigned dim0, unsigned dim1,
                                                   unsigned long long stride0)
{
  const unsigned long long ga = (unsigned long long)gaddr;
  v4u g0;
  g0[0] = 1u;                                   // count=1 valid user descriptor
  g0[1] = lds_addr;                             // lds_addr (bytes)
  g0[2] = (unsigned)(ga & 0xFFFFFFFFull);       // global_addr[31:0]
  g0[3] = (unsigned)((ga >> 32) & 0x01FFFFFFull) | 0x80000000u;  // addr[56:32] | type=2

  v8i g1;
  g1[0] = (int)(1u << 16);                                      // data_size = 2 bytes
  g1[1] = (int)((dim0 & 0xFFFFu) << 16);                        // tensor_dim0[15:0]
  g1[2] = (int)(((dim0 >> 16) & 0xFFFFu) | ((dim1 & 0xFFFFu) << 16));
  g1[3] = (int)(((dim1 >> 16) & 0xFFFFu) | (tile0 << 16));      // tile_dim0
  g1[4] = (int)(tile1 & 0xFFFFu);                               // tile_dim1 (tile_dim2=0)
  g1[5] = (int)(unsigned)(stride0 & 0xFFFFFFFFull);             // tensor_dim0_stride lo
  g1[6] = (int)(unsigned)((stride0 >> 32) & 0xFFFFull);         // stride hi (stride1=0)
  g1[7] = 0;

  v4i z4 = {0, 0, 0, 0};
#if __has_include(<hip/amd_detail/amd_gfx1250_TDM.h>)
  v8i z8 = {0, 0, 0, 0, 0, 0, 0, 0};            // amdgpu-toolchain: 6-arg form
  __builtin_amdgcn_tensor_load_to_lds(g0, g1, z4, z4, z8, 0);
#else
  __builtin_amdgcn_tensor_load_to_lds(g0, g1, z4, z4, 0);       // ROCm 7.2: 5-arg form
#endif
}
#endif

// ---------------------------------------------------------------------------
// Generic bf16 GEMM: C[M x Nn] = A[M x K] @ B[K x Nn] + bias, optional ReLU.
// B is supplied PRE-TRANSPOSED (Bt[Nn][K], row-major) so both global and LDS
// sides of the stage are contiguous; the 64x32 Bt tile is DMA'd by the TDM.
// Block = 256 threads = 8 waves; block tile 128(M) x 64(N); wave tile 16 x 64.
// ---------------------------------------------------------------------------
__global__ void k_gemm(const bf16_t* __restrict__ A, int lda,
                       const bf16_t* __restrict__ Bt,   // [Nn][K]
                       const float* __restrict__ bias,
                       float* __restrict__ Cf, bf16_t* __restrict__ Cb, int ldc,
                       int K, int Nn, int relu)
{
  __shared__ __attribute__((aligned(16))) bf16_t BsT[64 * 32];  // [n_local][k_local]
  const int m0   = blockIdx.y * 128;
  const int n0   = blockIdx.x * 64;
  const int t    = threadIdx.x;
  const int w    = t >> 5;                 // wave id 0..7 -> 16-row strip
  const int lane = t & 31;
  const int hw   = lane >> 4;              // half-wave select
  const int l16  = lane & 15;

  CFrag acc[4];
  #pragma unroll
  for (int j = 0; j < 4; ++j)
    #pragma unroll
    for (int r = 0; r < 8; ++r) acc[j].f[r] = 0.f;

  const bf16_t* Arow  = A  + (size_t)(m0 + w * 16 + l16) * lda;
  const bf16_t* Btile = Bt + (size_t)n0 * K;
#if defined(STAGE_TDM)
  const unsigned lbase = lds_off(BsT);
#endif

  for (int k0 = 0; k0 < K; k0 += 32) {
#if defined(STAGE_TDM)
    if (w == 0)                              // one wave drives the Tensor Data Mover
      tdm_load_2d(lbase, Btile + k0, /*tile0=*/32, /*tile1=*/64,
                  /*dim0=*/(unsigned)K, /*dim1=*/(unsigned)Nn,
                  /*stride0=*/(unsigned long long)K);
    __builtin_amdgcn_s_wait_tensorcnt(0);
    __syncthreads();
#elif defined(STAGE_ASYNC)
    {
      const int r = t >> 2, c = (t & 3) * 8;
      __builtin_amdgcn_global_load_async_to_lds_b128(
          (const __attribute__((address_space(1))) void*)(Btile + (size_t)r * K + k0 + c),
          (__attribute__((address_space(3))) void*)(BsT + r * 32 + c), 0, 0);
    }
    asm volatile("s_wait_asynccnt 0x0" ::: "memory");
    __syncthreads();
#else
    {
      const int r = t >> 2, c = (t & 3) * 8;
      *(uint4*)(BsT + r * 32 + c) = *(const uint4*)(Btile + (size_t)r * K + k0 + c);
    }
    __syncthreads();
#endif

    const int kb = hw ? 16 : 0;
    v16bf a = frag_split(Arow + k0 + kb);
    if (k0 + 32 < K) __builtin_prefetch(Arow + k0 + 32, 0, 1);  // global_prefetch_b8

    #pragma unroll
    for (int j = 0; j < 4; ++j) {
      const bf16_t* bp = BsT + (j * 16 + l16) * 32 + (hw ? 16 : 0);
      v16bf b = frag_contig(bp);                                // ds_load_b128 x2
      acc[j].v = wmma_bf16(a, b, acc[j].v);
    }
    __syncthreads();
  }

  // Epilogue (C/D layout: VGPR r -> M = r + 8*half-wave, N = lane&15)
  #pragma unroll
  for (int j = 0; j < 4; ++j) {
    const int col = n0 + j * 16 + l16;
    const float bv = bias ? bias[col] : 0.f;
    #pragma unroll
    for (int r = 0; r < 8; ++r) {
      const int row = m0 + w * 16 + r + (hw ? 8 : 0);
      float v = acc[j].f[r] + bv;
      if (relu) v = fmaxf(v, 0.f);
      if (Cf) Cf[(size_t)row * ldc + col] = v;
      if (Cb) Cb[(size_t)row * ldc + col] = (bf16_t)v;
    }
  }
}

// ---------------------------------------------------------------------------
// Causal attention, one workgroup per (head, 16-row query tile).
// Phase A: S = Q K^T * scale via WMMA, causal mask, bf16 scores into LDS.
// Phase B: fp32 row softmax (exp kept unnormalized, totals in s_tot).
// Phase C: ctx = P V via WMMA (A from LDS, B from pre-transposed V^T).
// ---------------------------------------------------------------------------
#define ATTN_SCALE 0.04419417382415922f   // 1/sqrt(512)

__global__ void k_attn(const bf16_t* __restrict__ Pq,  // projb [N][HD]
                       const bf16_t* __restrict__ Vt,  // V^T   [HD][N]
                       float* __restrict__ ctx)        // [N][HD]
{
  extern __shared__ bf16_t Pl[];          // [16][NEV] score/prob tile
  __shared__ float s_red[16][16];
  __shared__ float s_tot[16];

  const int qt   = blockIdx.x;
  const int h    = blockIdx.y;
  const int i0   = qt * 16;
  const int jmax = i0 + 16;               // causal key count (rows 0..i)
  const int jpad = (jmax + 31) & ~31;     // pad to WMMA K granularity
  const int t    = threadIdx.x;
  const int w    = t >> 5;
  const int lane = t & 31;
  const int hw   = lane >> 4;
  const int l16  = lane & 15;
  const bf16_t* Qh = Pq + h * DDIM;

  // Phase A: score tiles distributed over the 8 waves
  for (int jt = w; jt * 16 < jmax; jt += 8) {
    CFrag acc;
    #pragma unroll
    for (int r = 0; r < 8; ++r) acc.f[r] = 0.f;
    const bf16_t* Arow = Qh + (size_t)(i0 + l16) * HD;
    const bf16_t* Brow = Qh + (size_t)(jt * 16 + l16) * HD;
    for (int d0 = 0; d0 < DDIM; d0 += 32) {
      v16bf a = frag_split(Arow + d0 + (hw ? 16 : 0));
      v16bf b = frag_contig(Brow + d0 + (hw ? 16 : 0));  // K rows are contiguous d
      acc.v = wmma_bf16(a, b, acc.v);
    }
    #pragma unroll
    for (int r = 0; r < 8; ++r) {
      const int i = i0 + r + (hw ? 8 : 0);
      const int j = jt * 16 + l16;
      float s = acc.f[r] * ATTN_SCALE;
      if (j > i) s = -1e30f;              // causal mask
      Pl[(i - i0) * NEV + j] = (bf16_t)s;
    }
  }
  __syncthreads();

  // Phase B: softmax over each of the 16 rows (16 threads per row)
  {
    const int row = t >> 4, sub = t & 15;
    float m = -1e30f;
    for (int j = sub; j < jmax; j += 16) m = fmaxf(m, (float)Pl[row * NEV + j]);
    s_red[row][sub] = m;
    __syncthreads();
    m = -1e30f;
    #pragma unroll
    for (int k = 0; k < 16; ++k) m = fmaxf(m, s_red[row][k]);
    __syncthreads();
    float sum = 0.f;
    for (int j = sub; j < jmax; j += 16) {
      float pv = __expf((float)Pl[row * NEV + j] - m);
      Pl[row * NEV + j] = (bf16_t)pv;
      sum += pv;
    }
    for (int j = jmax + sub; j < jpad; j += 16) Pl[row * NEV + j] = (bf16_t)0.f;
    s_red[row][sub] = sum;
    __syncthreads();
    if (sub == 0) {
      float tot = 0.f;
      #pragma unroll
      for (int k = 0; k < 16; ++k) tot += s_red[row][k];
      s_tot[row] = tot;
    }
  }
  __syncthreads();

  // Phase C: each wave owns a 64-wide d-chunk of the output
  {
    const int dc0 = w * 64;
    CFrag acc[4];
    #pragma unroll
    for (int j = 0; j < 4; ++j)
      #pragma unroll
      for (int r = 0; r < 8; ++r) acc[j].f[r] = 0.f;

    for (int j0 = 0; j0 < jpad; j0 += 32) {
      v16bf a = frag_split(Pl + l16 * NEV + j0 + (hw ? 16 : 0));   // ds_load_b128 x2
      #pragma unroll
      for (int jj = 0; jj < 4; ++jj) {
        const bf16_t* vp = Vt + (size_t)(h * DDIM + dc0 + jj * 16 + l16) * NEV
                              + j0 + (hw ? 16 : 0);
        v16bf b = frag_contig(vp);
        acc[jj].v = wmma_bf16(a, b, acc[jj].v);
      }
    }
    #pragma unroll
    for (int jj = 0; jj < 4; ++jj)
      #pragma unroll
      for (int r = 0; r < 8; ++r) {
        const int li = r + (hw ? 8 : 0);
        const int d  = dc0 + jj * 16 + l16;
        ctx[(size_t)(i0 + li) * HD + h * DDIM + d] = acc[jj].f[r] / s_tot[li];
      }
  }
}

// ---------------------------------------------------------------------------
// projb [N][HD] -> Vt [HD][N] (32x32 LDS tiles)
// ---------------------------------------------------------------------------
__global__ void k_transpose(const bf16_t* __restrict__ in, bf16_t* __restrict__ out)
{
  __shared__ bf16_t tile[32][33];
  const int c0 = blockIdx.x * 32;
  const int n0 = blockIdx.y * 32;
  const int t  = threadIdx.x;
  const int r  = t >> 3;
  const int q  = (t & 7) * 4;
  #pragma unroll
  for (int i = 0; i < 4; ++i)
    tile[q + i][r] = in[(size_t)(n0 + r) * HD + c0 + q + i];
  __syncthreads();
  #pragma unroll
  for (int i = 0; i < 4; ++i)
    out[(size_t)(c0 + r) * NEV + n0 + q + i] = tile[r][q + i];
}

// ---------------------------------------------------------------------------
// LayerNorm kernels (fp32 stats, eps = 1e-5)
// ---------------------------------------------------------------------------
__global__ void k_ln1(const float* __restrict__ ctx, const float* __restrict__ xres,
                      const float* __restrict__ g, const float* __restrict__ b,
                      bf16_t* __restrict__ out)   // per-head residual LN -> bf16 cat
{
  __shared__ float r1[256], r2[256];
  const int n = blockIdx.x / HNUM;
  const int h = blockIdx.x % HNUM;
  const int t = threadIdx.x;
  const float* cp = ctx + (size_t)n * HD + h * DDIM;
  const float* xp = xres + (size_t)n * DDIM;
  const float y0 = cp[t] + xp[t];
  const float y1 = cp[t + 256] + xp[t + 256];
  r1[t] = y0 + y1;
  r2[t] = y0 * y0 + y1 * y1;
  __syncthreads();
  for (int s = 128; s > 0; s >>= 1) {
    if (t < s) { r1[t] += r1[t + s]; r2[t] += r2[t + s]; }
    __syncthreads();
  }
  const float mean = r1[0] / DDIM;
  const float var  = r2[0] / DDIM - mean * mean;
  const float rstd = rsqrtf(var + 1e-5f);
  bf16_t* op = out + (size_t)n * HD + h * DDIM;
  op[t]       = (bf16_t)((y0 - mean) * rstd * g[t]       + b[t]);
  op[t + 256] = (bf16_t)((y1 - mean) * rstd * g[t + 256] + b[t + 256]);
}

__global__ void k_ln2(const float* __restrict__ a, const float* __restrict__ br,
                      const float* __restrict__ g, const float* __restrict__ b,
                      float* __restrict__ out)    // final residual LN -> fp32
{
  __shared__ float r1[256], r2[256];
  const int n = blockIdx.x;
  const int t = threadIdx.x;
  const float* ap = a  + (size_t)n * DDIM;
  const float* bp = br + (size_t)n * DDIM;
  const float y0 = ap[t] + bp[t];
  const float y1 = ap[t + 256] + bp[t + 256];
  r1[t] = y0 + y1;
  r2[t] = y0 * y0 + y1 * y1;
  __syncthreads();
  for (int s = 128; s > 0; s >>= 1) {
    if (t < s) { r1[t] += r1[t + s]; r2[t] += r2[t + s]; }
    __syncthreads();
  }
  const float mean = r1[0] / DDIM;
  const float var  = r2[0] / DDIM - mean * mean;
  const float rstd = rsqrtf(var + 1e-5f);
  float* op = out + (size_t)n * DDIM;
  op[t]       = (y0 - mean) * rstd * g[t]       + b[t];
  op[t + 256] = (y1 - mean) * rstd * g[t + 256] + b[t + 256];
}

// ---------------------------------------------------------------------------
// Utilities
// ---------------------------------------------------------------------------
__global__ void k_copy_f(const float* __restrict__ in, float* __restrict__ out, int n)
{
  const int i = blockIdx.x * 256 + threadIdx.x;
  if (i < n) out[i] = in[i];
}

__global__ void k_cvt(const float* __restrict__ in, bf16_t* __restrict__ out, int n)
{
  const int i = blockIdx.x * 256 + threadIdx.x;
  if (i < n) out[i] = (bf16_t)in[i];
}

// fp32 [K][N] -> bf16 transposed [N][K]
__global__ void k_cvt_t(const float* __restrict__ in, bf16_t* __restrict__ out,
                        int K, int N)
{
  const int i = blockIdx.x * 256 + threadIdx.x;  // over N*K
  const int n = i / K, k = i % K;
  out[i] = (bf16_t)in[(size_t)k * N + n];
}

// tile(x,(1,H)) @ Wh == x @ (sum_h Wh[h*D:(h+1)*D, :]); fold + transpose:
// out[j*D + k] = sum_h Wh[(h*D+k)*HD + j]   (out is [HD][D] bf16)
__global__ void k_reduce_wh_t(const float* __restrict__ Wh, bf16_t* __restrict__ out)
{
  const int i = blockIdx.x * 256 + threadIdx.x;   // over HD*DDIM
  const int j = i / DDIM, k = i % DDIM;
  float s = 0.f;
  #pragma unroll
  for (int hh = 0; hh < HNUM; ++hh)
    s += Wh[(size_t)(hh * DDIM + k) * HD + j];
  out[i] = (bf16_t)s;
}

// ---------------------------------------------------------------------------
extern "C" void kernel_launch(void* const* d_in, const int* in_sizes, int n_in,
                              void* d_out, int out_size, void* d_ws, size_t ws_size,
                              hipStream_t stream)
{
  (void)in_sizes; (void)n_in; (void)out_size; (void)ws_size;
  const float* x     = (const float*)d_in[0];
  const float* Wh    = (const float*)d_in[1];
  const float* bh    = (const float*)d_in[2];
  const float* Wr    = (const float*)d_in[3];
  const float* br    = (const float*)d_in[4];
  const float* W1    = (const float*)d_in[5];
  const float* b1    = (const float*)d_in[6];
  const float* W2    = (const float*)d_in[7];
  const float* b2    = (const float*)d_in[8];
  const float* gamma = (const float*)d_in[9];
  const float* beta  = (const float*)d_in[10];
  float* out = (float*)d_out;

  char* p = (char*)d_ws;
  auto alloc = [&](size_t bytes) { char* r = p; p += (bytes + 255) & ~(size_t)255; return r; };
  float*  xcur  = (float*) alloc((size_t)NEV * DDIM * 4);
  bf16_t* xb    = (bf16_t*)alloc((size_t)NEV * DDIM * 2);
  bf16_t* bWh   = (bf16_t*)alloc((size_t)HD * DDIM * 2);   // [HD][D]  (Wh_eff^T)
  bf16_t* bWr   = (bf16_t*)alloc((size_t)DDIM * HD * 2);   // [D][HD]  (Wr^T)
  bf16_t* bW1   = (bf16_t*)alloc((size_t)DDIM * DDIM * 2); // [D][D]   (W1^T)
  bf16_t* bW2   = (bf16_t*)alloc((size_t)DDIM * DDIM * 2); // [D][D]   (W2^T)
  bf16_t* projb = (bf16_t*)alloc((size_t)NEV * HD * 2);
  bf16_t* Vt    = (bf16_t*)alloc((size_t)HD * NEV * 2);
  float*  ctx   = (float*) alloc((size_t)NEV * HD * 4);
  bf16_t* catb  = (bf16_t*)alloc((size_t)NEV * HD * 2);
  float*  asub  = (float*) alloc((size_t)NEV * DDIM * 4);
  bf16_t* asubb = (bf16_t*)alloc((size_t)NEV * DDIM * 2);
  bf16_t* ff1b  = (bf16_t*)alloc((size_t)NEV * DDIM * 2);
  float*  ff2   = (float*) alloc((size_t)NEV * DDIM * 4);

  k_copy_f<<<NEV * DDIM / 256, 256, 0, stream>>>(x, xcur, NEV * DDIM);

  for (int l = 0; l < LNUM; ++l) {
    // Per-layer weight prep: fold Wh over heads, transpose everything to [Nn][K] bf16.
    k_reduce_wh_t<<<HD * DDIM / 256, 256, 0, stream>>>(Wh + (size_t)l * HD * HD, bWh);
    k_cvt_t<<<DDIM * HD / 256, 256, 0, stream>>>(Wr + (size_t)l * HD * DDIM, bWr, HD, DDIM);
    k_cvt_t<<<DDIM * DDIM / 256, 256, 0, stream>>>(W1 + (size_t)l * DDIM * DDIM, bW1, DDIM, DDIM);
    k_cvt_t<<<DDIM * DDIM / 256, 256, 0, stream>>>(W2 + (size_t)l * DDIM * DDIM, bW2, DDIM, DDIM);
    k_cvt<<<NEV * DDIM / 256, 256, 0, stream>>>(xcur, xb, NEV * DDIM);

    // proj = x @ Wh_eff + bh  -> bf16 [N][HD]
    k_gemm<<<dim3(HD / 64, NEV / 128), 256, 0, stream>>>(
        xb, DDIM, bWh, bh + (size_t)l * HD, nullptr, projb, HD, DDIM, HD, 0);
    k_transpose<<<dim3(HD / 32, NEV / 32), 256, 0, stream>>>(projb, Vt);
    k_attn<<<dim3(NEV / 16, HNUM), 256, 16 * NEV * 2, stream>>>(projb, Vt, ctx);
    k_ln1<<<NEV * HNUM, 256, 0, stream>>>(
        ctx, xcur, gamma + (size_t)l * DDIM, beta + (size_t)l * DDIM, catb);
    // attn_sub = cat @ Wr + br
    k_gemm<<<dim3(DDIM / 64, NEV / 128), 256, 0, stream>>>(
        catb, HD, bWr, br + (size_t)l * DDIM, asub, asubb, DDIM, HD, DDIM, 0);
    // ff1 = relu(attn_sub @ W1 + b1)
    k_gemm<<<dim3(DDIM / 64, NEV / 128), 256, 0, stream>>>(
        asubb, DDIM, bW1, b1 + (size_t)l * DDIM, nullptr, ff1b, DDIM, DDIM, DDIM, 1);
    // ff2 = ff1 @ W2 + b2
    k_gemm<<<dim3(DDIM / 64, NEV / 128), 256, 0, stream>>>(
        ff1b, DDIM, bW2, b2 + (size_t)l * DDIM, ff2, nullptr, DDIM, DDIM, DDIM, 0);

    float* dst = (l == LNUM - 1) ? out : xcur;
    k_ln2<<<NEV, 256, 0, stream>>>(
        ff2, asub, gamma + (size_t)l * DDIM, beta + (size_t)l * DDIM, dst);
  }
}
